// NekoMindMoeTopKRouter_30614526886227
// MI455X (gfx1250) — matrix operand order repaired and checked by reading, not compiled, not run on hardware
//
#include <hip/hip_runtime.h>

// MoE top-k router for MI455X (gfx1250, wave32, WMMA).
// GEMM: x[16384,4096] * w[64,4096]^T -> logits[16384,64] via fp16x3 emulation:
// A,B split into f16 hi + f16 lo (v_cvt_pkrtz), accumulate
// Ahi*Bhi + Ahi*Blo + Alo*Bhi in fp32 with v_wmma_f32_16x16x32_f16.
// Then per-row top-8 + renormalized softmax over the 8 winners only
// (the full softmax denominator cancels in normalized top-k probabilities).

typedef _Float16 v16h __attribute__((ext_vector_type(16)));
typedef __fp16   pk2  __attribute__((ext_vector_type(2)));   // cvt_pkrtz result type
typedef float    v8f  __attribute__((ext_vector_type(8)));

#define T_TOKENS 16384
#define H_DIM    4096
#define E_EXP    64
#define TOPK     8

#define WAVES_PER_BLOCK 4
#define ROWS_PER_BLOCK  64            // 16 rows per wave
#define KCHUNK          64            // two 32-wide WMMA sub-chunks per iteration
#define LOG_STRIDE      68            // padded LDS row stride -> conflict-free b128 reads

// hi = rtz_f16(a), lo = rtz_f16(a - hi): one v_cvt_pkrtz per pair per term
__device__ __forceinline__ void f16split2(float a, float b, pk2 &hi, pk2 &lo) {
    pk2 h = __builtin_amdgcn_cvt_pkrtz(a, b);
    float ra = a - (float)h.x;
    float rb = b - (float)h.y;
    hi = h;
    lo = __builtin_amdgcn_cvt_pkrtz(ra, rb);
}

union FragH {
    pk2  h[8];
    v16h v;
};

union PackH {
    pk2   h[8];
    uint4 q[2];
};

__global__ __launch_bounds__(128, 1)
void moe_router_kernel(const float* __restrict__ x,
                       const float* __restrict__ w,
                       float* __restrict__ out_logits,
                       float* __restrict__ out_scores,
                       float* __restrict__ out_indices)
{
    // B fragments for one 64-wide K chunk (two 32-wide sub-chunks), all 64
    // experts, f16 hi+lo. Per sub-chunk s, expert e: elems [h*16+j] =
    // f16(w[e][kc + 32*s + 16*h + j])  (B v16h element j <-> K = 16*half + j)
    __shared__ __align__(32) unsigned short sBhi[2 * E_EXP * 32];
    __shared__ __align__(32) unsigned short sBlo[2 * E_EXP * 32];
    __shared__ __align__(16) float sLog[WAVES_PER_BLOCK * 16 * LOG_STRIDE];

    const int tid    = threadIdx.x;
    const int lane   = tid & 31;
    const int wave   = tid >> 5;
    const int laneLo = lane & 15;
    const int hl     = lane >> 4;          // lane half (0/1)

    const int rowBase = blockIdx.x * ROWS_PER_BLOCK + wave * 16;
    const int myRow   = rowBase + laneLo;

    // cooperative-B mapping: 128 threads, 2 per expert, 16 contiguous K each
    // (per sub-chunk); thread handles both sub-chunks of the 64-wide chunk.
    const int be = tid >> 1;               // expert 0..63
    const int bh = tid & 1;                // frag half (K 0-15 vs 16-31 of sub-chunk)
    const float* wSrc = w + (size_t)be * H_DIM + bh * 16;
    const int bDst = be * 32 + bh * 16;    // ushort offset within a sub-chunk slab

    // A source: contiguous layout -> lane reads K = base..base+15
    // with base = 16*half, relative to sub-chunk start.
    const float* aSrc = x + (size_t)myRow * H_DIM + 16 * hl;

    v8f acc[4];
    const v8f vzero = {0.f, 0.f, 0.f, 0.f, 0.f, 0.f, 0.f, 0.f};
#pragma unroll
    for (int t = 0; t < 4; ++t) acc[t] = vzero;

    const int bReadBase = laneLo * 32 + hl * 16;   // + 512*t per tile, + 2048*s

    for (int kc = 0; kc < H_DIM; kc += KCHUNK) {
        __syncthreads();   // protect LDS slab from previous iteration's readers

        // prefetch next A chunk (speculative; clamped at the tail)
        {
            const float* pf = aSrc + (kc + KCHUNK < H_DIM ? kc + KCHUNK : kc);
            __builtin_prefetch(pf, 0, 3);
        }

        // ---- cooperative weight chunk: fp32 -> f16 hi/lo into LDS ----
#pragma unroll
        for (int s = 0; s < 2; ++s) {
            const float* src = wSrc + kc + 32 * s;
            float4 q0 = *(const float4*)(src + 0);
            float4 q1 = *(const float4*)(src + 4);
            float4 q2 = *(const float4*)(src + 8);
            float4 q3 = *(const float4*)(src + 12);
            float bf[16] = {q0.x, q0.y, q0.z, q0.w, q1.x, q1.y, q1.z, q1.w,
                            q2.x, q2.y, q2.z, q2.w, q3.x, q3.y, q3.z, q3.w};
            PackH uh, ul;
#pragma unroll
            for (int i = 0; i < 8; ++i)
                f16split2(bf[2 * i], bf[2 * i + 1], uh.h[i], ul.h[i]);
            unsigned short* dh = &sBhi[s * 2048 + bDst];
            unsigned short* dl = &sBlo[s * 2048 + bDst];
            *(uint4*)(dh)     = uh.q[0];
            *(uint4*)(dh + 8) = uh.q[1];
            *(uint4*)(dl)     = ul.q[0];
            *(uint4*)(dl + 8) = ul.q[1];
        }
        __syncthreads();

        // ---- per-wave A fragments (two sub-chunks), f16 hi/lo split ----
        FragH ahi[2], alo[2];
#pragma unroll
        for (int s = 0; s < 2; ++s) {
            const float* src = aSrc + kc + 32 * s;
            float4 a0 = *(const float4*)(src + 0);
            float4 a1 = *(const float4*)(src + 4);
            float4 a2 = *(const float4*)(src + 8);
            float4 a3 = *(const float4*)(src + 12);
            float av[16] = {a0.x, a0.y, a0.z, a0.w, a1.x, a1.y, a1.z, a1.w,
                            a2.x, a2.y, a2.z, a2.w, a3.x, a3.y, a3.z, a3.w};
#pragma unroll
            for (int i = 0; i < 8; ++i)
                f16split2(av[2 * i], av[2 * i + 1], ahi[s].h[i], alo[s].h[i]);
        }

        // ---- 2 sub-chunks x 4 expert tiles x 3 WMMA terms ----
#pragma unroll
        for (int s = 0; s < 2; ++s) {
#pragma unroll
            for (int t = 0; t < 4; ++t) {
                const int boff = s * 2048 + t * 512 + bReadBase;
                v16h bhiV = *(const v16h*)&sBhi[boff];
                v16h bloV = *(const v16h*)&sBlo[boff];
                acc[t] = __builtin_amdgcn_wmma_f32_16x16x32_f16(
                    false, ahi[s].v, false, bhiV, (short)0, acc[t], false, false);
                acc[t] = __builtin_amdgcn_wmma_f32_16x16x32_f16(
                    false, ahi[s].v, false, bloV, (short)0, acc[t], false, false);
                acc[t] = __builtin_amdgcn_wmma_f32_16x16x32_f16(
                    false, alo[s].v, false, bhiV, (short)0, acc[t], false, false);
            }
        }
    }

    // ---- write raw logits (coalesced) + stash in LDS for top-k ----
    float* lbase = &sLog[wave * 16 * LOG_STRIDE];
#pragma unroll
    for (int t = 0; t < 4; ++t) {
#pragma unroll
        for (int r = 0; r < 8; ++r) {
            // C/D layout: VGPR r -> row r (lanes 0-15) / row r+8 (lanes 16-31)
            const int row = 8 * hl + r;
            const int col = 16 * t + laneLo;
            const float v = acc[t][r];
            out_logits[(size_t)(rowBase + row) * E_EXP + col] = v;
            lbase[row * LOG_STRIDE + col] = v;
        }
    }
    __syncthreads();

    // ---- top-8 + renormalized softmax over winners (lanes 0-15, 1 row each) ----
    if (lane < 16) {
        const float* lr = &lbase[lane * LOG_STRIDE];
        float v[64];
#pragma unroll
        for (int j = 0; j < 16; ++j) {
            float4 q = *(const float4*)&lr[4 * j];
            v[4 * j + 0] = q.x; v[4 * j + 1] = q.y;
            v[4 * j + 2] = q.z; v[4 * j + 3] = q.w;
        }

        unsigned long long used = 0ULL;
        float topv[TOPK];
        int   topi[TOPK];
#pragma unroll
        for (int k = 0; k < TOPK; ++k) {
            float best = -3.0e38f;
            int   bi   = 0;
#pragma unroll
            for (int j = 0; j < 64; ++j) {
                const bool freeSlot = (((used >> j) & 1ULL) == 0ULL);
                if (freeSlot && (v[j] > best)) { best = v[j]; bi = j; }
            }
            used |= (1ULL << bi);
            topv[k] = best;
            topi[k] = bi;
        }

        // normalized top-k softmax: exp(l_i - max) / sum over the 8 winners
        const float m = topv[0];
        float e[TOPK];
        float s = 0.f;
#pragma unroll
        for (int k = 0; k < TOPK; ++k) { e[k] = __expf(topv[k] - m); s += e[k]; }
        const float inv = 1.0f / s;

        const size_t ob = (size_t)(rowBase + lane) * TOPK;
#pragma unroll
        for (int k = 0; k < TOPK; ++k) {
            out_scores[ob + k]  = e[k] * inv;
            out_indices[ob + k] = (float)topi[k];
        }
    }
}

extern "C" void kernel_launch(void* const* d_in, const int* in_sizes, int n_in,
                              void* d_out, int out_size, void* d_ws, size_t ws_size,
                              hipStream_t stream) {
    const float* x = (const float*)d_in[0];   // [16384, 4096] fp32
    const float* w = (const float*)d_in[1];   // [64, 4096] fp32
    float* out = (float*)d_out;
    float* out_logits  = out;                                   // [T, 64]
    float* out_scores  = out + (size_t)T_TOKENS * E_EXP;        // [T, 8]
    float* out_indices = out_scores + (size_t)T_TOKENS * TOPK;  // [T, 8] (as float)

    dim3 grid(T_TOKENS / ROWS_PER_BLOCK);   // 256 blocks
    dim3 block(32 * WAVES_PER_BLOCK);       // 128 threads = 4 wave32s
    hipLaunchKernelGGL(moe_router_kernel, grid, block, 0, stream,
                       x, w, out_logits, out_scores, out_indices);
}
